// GCNEncoder_84619445665914
// MI455X (gfx1250) — matrix operand order repaired and checked by reading, not compiled, not run on hardware
//
#include <hip/hip_runtime.h>
#include <hip/hip_bf16.h>
#include <stdint.h>

// ---------- types ----------
typedef __attribute__((ext_vector_type(16))) __bf16          bf16x16;
typedef __attribute__((ext_vector_type(8)))  float           f32x8;
typedef __attribute__((ext_vector_type(16))) unsigned short  u16x16;

__device__ __forceinline__ unsigned short f2bf(float f) {
    unsigned u = __builtin_bit_cast(unsigned, f);
    unsigned r = u + 0x7FFFu + ((u >> 16) & 1u);   // round-to-nearest-even
    return (unsigned short)(r >> 16);
}
__device__ __forceinline__ float bf2f(unsigned short h) {
    unsigned u = ((unsigned)h) << 16;
    return __builtin_bit_cast(float, u);
}

// ---------- degree / norm ----------
__global__ void k_deg_init(float* deg, int n) {
    int i = blockIdx.x * blockDim.x + threadIdx.x;
    if (i < n) deg[i] = 1.0f;                       // self-loop
}
__global__ void k_deg_accum(const int* __restrict__ dst, float* deg, int E) {
    int e = blockIdx.x * blockDim.x + threadIdx.x;
    if (e < E) atomicAdd(&deg[dst[e]], 1.0f);
}
__global__ void k_deg_rsqrt(float* deg, int n) {
    int i = blockIdx.x * blockDim.x + threadIdx.x;
    if (i < n) deg[i] = rsqrtf(deg[i]);             // deg >= 1 always
}

// ---------- fp32 -> bf16 bulk convert (4 elems / thread) ----------
__global__ void k_cvt_bf16(const float* __restrict__ in, unsigned short* __restrict__ out, size_t n4) {
    size_t i = (size_t)blockIdx.x * blockDim.x + threadIdx.x;
    if (i >= n4) return;
    const float4 v = *(const float4*)(in + i * 4);
    uint2 o;
    o.x = (unsigned)f2bf(v.x) | ((unsigned)f2bf(v.y) << 16);
    o.y = (unsigned)f2bf(v.z) | ((unsigned)f2bf(v.w) << 16);
    *(uint2*)(out + i * 4) = o;
}

// ---------- pack W[K=256][nout] into per-lane WMMA B-fragment order ----------
// Wp index = ((kt*ntiles + ntg)*32 + lane)*16 + i
// element: K = kt*32 + (lane>>4)*8 + (i&7) + (i>>3)*16 ; N = ntg*16 + (lane&15)
__global__ void k_pack_w(const float* __restrict__ W, unsigned short* __restrict__ Wp, int nout) {
    int p = blockIdx.x * blockDim.x + threadIdx.x;
    int total = 256 * nout;
    if (p >= total) return;
    int i    = p & 15;
    int L    = (p >> 4) & 31;
    int rest = p >> 9;
    int ntiles = nout >> 4;
    int ntg  = rest % ntiles;
    int kt   = rest / ntiles;
    int k = kt * 32 + (L >> 4) * 8 + (i & 7) + (i >> 3) * 16;
    int n = ntg * 16 + (L & 15);
    Wp[p] = f2bf(W[(size_t)k * nout + n]);
}

// ---------- WMMA GEMM: Out[rows,NOUT](bf16) = A[rows,256](bf16) @ Wp ----------
// 128 threads = 4 waves; each wave computes a 32 x (16*NTPW) tile
// (2 row tiles so every B fragment feeds 2 back-to-back WMMAs).
template<int NOUT, int NTPW>
__global__ __launch_bounds__(128) void k_gemm_wmma(
    const unsigned short* __restrict__ A,
    const unsigned short* __restrict__ Wp,
    unsigned short* __restrict__ Out,
    int rows)
{
    const int lane = threadIdx.x & 31;
    const int wave = threadIdx.x >> 5;
    const int half = lane >> 4;      // K-half selector
    const int m16  = lane & 15;
    const int rowTile0 = blockIdx.x * 2;
    constexpr int NTILES = NOUT / 16;

    int row0 = rowTile0 * 16 + m16;
    int row1 = row0 + 16;
    int r0c = row0 < rows ? row0 : rows - 1;
    int r1c = row1 < rows ? row1 : rows - 1;
    const unsigned short* arow0 = A + (size_t)r0c * 256;
    const unsigned short* arow1 = A + (size_t)r1c * 256;

    f32x8 acc[2][NTPW];
#pragma unroll
    for (int m = 0; m < 2; ++m)
#pragma unroll
        for (int t = 0; t < NTPW; ++t) acc[m][t] = (f32x8)(0.0f);

#pragma unroll 2
    for (int kt = 0; kt < 8; ++kt) {
        const int kbase = kt * 32 + half * 8;

        // A fragments for both row tiles: 16-bit A 16x32 layout,
        // two contiguous 16B loads per lane each.
        union { u16x16 v; uint4 q[2]; } a0u, a1u;
        a0u.q[0] = *(const uint4*)(arow0 + kbase);
        a0u.q[1] = *(const uint4*)(arow0 + kbase + 16);
        a1u.q[0] = *(const uint4*)(arow1 + kbase);
        a1u.q[1] = *(const uint4*)(arow1 + kbase + 16);
        const bf16x16 af0 = __builtin_bit_cast(bf16x16, a0u.v);
        const bf16x16 af1 = __builtin_bit_cast(bf16x16, a1u.v);

        // Load ALL B fragments for this k-step first (one load clause),
        // then issue the WMMA burst.
        bf16x16 bfv[NTPW];
#pragma unroll
        for (int t = 0; t < NTPW; ++t) {
            const int ntg = wave * NTPW + t;
            const unsigned short* wptr =
                Wp + ((((size_t)kt * NTILES + ntg) * 32) + lane) * 16;
            union { u16x16 v; uint4 q[2]; } bu;
            bu.q[0] = *(const uint4*)(wptr);
            bu.q[1] = *(const uint4*)(wptr + 8);
            bfv[t] = __builtin_bit_cast(bf16x16, bu.v);
        }
#pragma unroll
        for (int t = 0; t < NTPW; ++t) {
            acc[0][t] = __builtin_amdgcn_wmma_f32_16x16x32_bf16(
                false, af0, false, bfv[t], (short)0, acc[0][t], false, false);
            acc[1][t] = __builtin_amdgcn_wmma_f32_16x16x32_bf16(
                false, af1, false, bfv[t], (short)0, acc[1][t], false, false);
        }
    }

    // C/D layout: N = lane&15 ; M = v + 8*(lane>>4)
#pragma unroll
    for (int m = 0; m < 2; ++m) {
#pragma unroll
        for (int t = 0; t < NTPW; ++t) {
            const int col = (wave * NTPW + t) * 16 + m16;
#pragma unroll
            for (int v = 0; v < 8; ++v) {
                const int r = (rowTile0 + m) * 16 + half * 8 + v;
                if (r < rows) Out[(size_t)r * NOUT + col] = f2bf(acc[m][t][v]);
            }
        }
    }
}

// ---------- bias broadcast init (C must be a power of two) ----------
__global__ void k_init_bias(float* __restrict__ out, const float* __restrict__ b,
                            int Cmask, size_t total) {
    size_t i = (size_t)blockIdx.x * blockDim.x + threadIdx.x;
    if (i < total) out[i] = b[i & Cmask];
}

// ---------- gather * norm -> atomic scatter (one wave per virtual edge) ----------
template<int C>
__global__ __launch_bounds__(256) void k_scatter(
    const unsigned short* __restrict__ h,
    const int* __restrict__ src, const int* __restrict__ dst,
    const float* __restrict__ dinv,
    float* __restrict__ agg,
    int E, int Nn)
{
    const int wid  = blockIdx.x * (blockDim.x >> 5) + (threadIdx.x >> 5);
    const int lane = threadIdx.x & 31;
    if (wid >= E + Nn) return;
    int s, d;
    if (wid < E) { s = src[wid]; d = dst[wid]; }
    else         { s = wid - E;  d = s; }          // self-loop
    const float w = dinv[s] * dinv[d];

    constexpr int CPL = C / 32;                    // 8 (C=256) or 4 (C=128)
    const unsigned short* hp = h   + (size_t)s * C + lane * CPL;
    float*                ap = agg + (size_t)d * C + lane * CPL;

    unsigned short buf[CPL];
    if constexpr (CPL == 8) { *(uint4*)buf = *(const uint4*)hp; }
    else                    { *(uint2*)buf = *(const uint2*)hp; }
#pragma unroll
    for (int i = 0; i < CPL; ++i)
        atomicAdd(ap + i, bf2f(buf[i]) * w);
}

// ---------- ReLU + convert to bf16 ----------
__global__ void k_relu_cvt(const float* __restrict__ in, unsigned short* __restrict__ out, size_t n) {
    size_t i = (size_t)blockIdx.x * blockDim.x + threadIdx.x;
    if (i < n) out[i] = f2bf(fmaxf(in[i], 0.0f));
}

// =====================================================================
extern "C" void kernel_launch(void* const* d_in, const int* in_sizes, int n_in,
                              void* d_out, int out_size, void* d_ws, size_t ws_size,
                              hipStream_t stream) {
    const float* x  = (const float*)d_in[0];
    const int*   ei = (const int*)  d_in[1];
    const float* W1 = (const float*)d_in[2];
    const float* b1 = (const float*)d_in[3];
    const float* W2 = (const float*)d_in[4];
    const float* b2 = (const float*)d_in[5];

    const int Nn = in_sizes[0] / 256;   // nodes
    const int E  = in_sizes[1] / 2;     // edges
    const int* src = ei;
    const int* dst = ei + E;

    // workspace carve-out
    char*  ws  = (char*)d_ws;
    size_t off = 0;
    auto alloc = [&](size_t bytes) -> void* {
        void* p = ws + off;
        off += (bytes + 255) & ~(size_t)255;
        return p;
    };
    float*          dinv = (float*)         alloc((size_t)Nn * 4);
    unsigned short* xb   = (unsigned short*)alloc((size_t)Nn * 256 * 2); // x bf16, later hb
    unsigned short* W1p  = (unsigned short*)alloc((size_t)256 * 256 * 2);
    unsigned short* W2p  = (unsigned short*)alloc((size_t)256 * 128 * 2);
    unsigned short* hbuf = (unsigned short*)alloc((size_t)Nn * 256 * 2); // GEMM outputs (bf16)
    float*          agg  = (float*)         alloc((size_t)Nn * 256 * 4); // fp32 aggregator

    const int T = 256;
    auto cdiv = [](long long a, long long b) { return (int)((a + b - 1) / b); };

    // 1) symmetric normalization
    k_deg_init <<<cdiv(Nn, T), T, 0, stream>>>(dinv, Nn);
    k_deg_accum<<<cdiv(E,  T), T, 0, stream>>>(dst, dinv, E);
    k_deg_rsqrt<<<cdiv(Nn, T), T, 0, stream>>>(dinv, Nn);

    // 2) input convert + weight packing
    {
        size_t n4 = (size_t)Nn * 256 / 4;
        k_cvt_bf16<<<cdiv((long long)n4, T), T, 0, stream>>>(x, xb, n4);
    }
    k_pack_w<<<cdiv(256 * 256, T), T, 0, stream>>>(W1, W1p, 256);
    k_pack_w<<<cdiv(256 * 128, T), T, 0, stream>>>(W2, W2p, 128);

    const int gemmBlocks = cdiv(Nn, 32);          // 2 row tiles per block
    const int waveItems  = E + Nn;                // edges + self-loops
    const int scatterBlocks = cdiv(waveItems, 8); // 8 waves / 256-thread block

    // 3) layer 1: h1 = xb @ W1  (bf16 out), aggregate with norm, +b1, ReLU
    k_gemm_wmma<256, 4><<<gemmBlocks, 128, 0, stream>>>(xb, W1p, hbuf, Nn);
    k_init_bias<<<cdiv((long long)Nn * 256, T), T, 0, stream>>>(agg, b1, 255, (size_t)Nn * 256);
    k_scatter<256><<<scatterBlocks, 256, 0, stream>>>(hbuf, src, dst, dinv, agg, E, Nn);
    k_relu_cvt<<<cdiv((long long)Nn * 256, T), T, 0, stream>>>(agg, xb, (size_t)Nn * 256);

    // 4) layer 2: h2 = hb @ W2 (bf16 out), aggregate into d_out (pre-filled with b2)
    k_gemm_wmma<128, 2><<<gemmBlocks, 128, 0, stream>>>(xb, W2p, hbuf, Nn);
    k_init_bias<<<cdiv((long long)Nn * 128, T), T, 0, stream>>>((float*)d_out, b2, 127, (size_t)Nn * 128);
    k_scatter<128><<<scatterBlocks, 256, 0, stream>>>(hbuf, src, dst, dinv, (float*)d_out, E, Nn);
}
